// Agent_64647847740216
// MI455X (gfx1250) — matrix operand order repaired and checked by reading, not compiled, not run on hardware
//
#include <hip/hip_runtime.h>

// ---------------------------------------------------------------------------
// CDNA5 (gfx1250) implementation: bf16 WMMA (f32 accum) for all GEMM-shaped
// work (FC, embedding, enc-LSTM gates, 4-layer mem-LSTM gates), fp32 VALU for
// the convolutions and heads. Wave32 throughout. The mem-LSTM weight stream is
// staged into LDS with the Tensor Data Mover (double-buffered, s_wait_tensorcnt).
// ---------------------------------------------------------------------------

typedef unsigned short u16;
typedef __attribute__((ext_vector_type(16))) __bf16 v16bf;
typedef __attribute__((ext_vector_type(8)))  float  v8f;
typedef __attribute__((ext_vector_type(4)))  unsigned int u32x4;
typedef __attribute__((ext_vector_type(8)))  int i32x8;
typedef __attribute__((ext_vector_type(4)))  int i32x4;

#define T_STEPS 64
#define BATCH   32
#define ENC_H   256
#define MEM_H   512
#define MEM_IN  288
#define EMB_D   32
#define NACT    8

#if defined(__has_builtin)
#  if __has_builtin(__builtin_amdgcn_tensor_load_to_lds) && __has_builtin(__builtin_amdgcn_s_wait_tensorcnt)
#    define HAVE_TDM 1
#  endif
#endif
#ifndef HAVE_TDM
#  define HAVE_TDM 0
#endif

// ---------------- scalar helpers ----------------
__device__ inline u16 f2bf(float f) {
  unsigned int u = __float_as_uint(f);
  unsigned int r = u + 0x7FFFu + ((u >> 16) & 1u);   // round-to-nearest-even
  return (u16)(r >> 16);
}
__device__ inline float bf2f(u16 h) { return __uint_as_float(((unsigned int)h) << 16); }
__device__ inline float sigm(float x) { return 1.f / (1.f + __expf(-x)); }

// ---------------- WMMA fragment helpers ----------------
__device__ inline v16bf make_frag(const u16* plo, const u16* phi) {
  union { v16bf v; u32x4 q[2]; } u;
  u.q[0] = *(const u32x4*)plo;
  u.q[1] = *(const u32x4*)phi;
  return u.v;
}
// A fragment: 16x32 bf16 tile from row-major matrix X with leading dim ld.
// lanes 0-15: M=lane, K = kb+0..7 (v0-3) and kb+16..23 (v4-7)
// lanes 16-31: M=lane-16, K = kb+8..15 and kb+24..31
__device__ inline v16bf frag_a(const u16* X, int ld, int rowBase, int kBase, int lane) {
  int row = rowBase + (lane & 15);
  int kg  = (lane >> 4) * 8;
  const u16* p = X + (size_t)row * ld + kBase + kg;
  return make_frag(p, p + 16);
}
// B fragment: pre-packed tiles [tile][lane][16] (lane n=lane&15, K = 8*(lane>>4)+{0..7,16..23})
__device__ inline v16bf frag_b(const u16* BP, int tile, int lane) {
  const u16* p = BP + (size_t)tile * 512 + lane * 16;
  return make_frag(p, p + 8);
}
__device__ inline v8f wmma_bf16(v16bf a, v16bf b, v8f c) {
  return __builtin_amdgcn_wmma_f32_16x16x32_bf16(false, a, false, b, (short)0, c, false, false);
}
__device__ inline v8f vzero8() { v8f a; for (int e = 0; e < 8; ++e) a[e] = 0.f; return a; }

#if HAVE_TDM
// TDM: copy 4 gate-runs of 8 B-fragment tiles (4 x 4096 bf16, row stride
// 32 tiles = 16384 elements) from global into LDS. 2D descriptor, data_size=2B.
__device__ __attribute__((always_inline)) inline
void tdm_issue(const u16* gptr, unsigned ldsAddr) {
  unsigned long long ga = (unsigned long long)(uintptr_t)gptr;
  u32x4 g0;
  g0[0] = 1u;                                      // count=1, user descriptor
  g0[1] = ldsAddr;                                 // lds_addr (bytes)
  g0[2] = (unsigned)(ga & 0xFFFFFFFFu);            // global_addr[31:0]
  g0[3] = (unsigned)((ga >> 32) & 0x1FFFFFFu) | 0x80000000u;  // addr[56:32] | type=2
  const unsigned D0 = 1u << 20;                    // tensor_dim0 (large: no OOB clip)
  i32x8 g1;
  g1[0] = 0x00010000;                              // wg_mask=0 | data_size=2B
  g1[1] = (int)((D0 & 0xFFFFu) << 16);             // abar=0 | tensor_dim0.lo16
  g1[2] = (int)((D0 >> 16) | (4u << 16));          // tensor_dim0.hi16 | tensor_dim1=4 .lo16
  g1[3] = (int)(4096u << 16);                      // tensor_dim1.hi16 | tile_dim0=4096
  g1[4] = 4;                                       // tile_dim1=4 | tile_dim2=0
  g1[5] = 16384;                                   // tensor_dim0_stride lo32 (elements)
  g1[6] = 0;                                       // stride hi16 | dim1_stride lo16
  g1[7] = 0;
  i32x4 z4; z4[0] = z4[1] = z4[2] = z4[3] = 0;
#if __clang_major__ >= 23
  i32x8 z8; for (int i = 0; i < 8; ++i) z8[i] = 0;
  __builtin_amdgcn_tensor_load_to_lds(g0, g1, z4, z4, z8, 0);
#else
  __builtin_amdgcn_tensor_load_to_lds(g0, g1, z4, z4, 0);
#endif
}
#endif

// ---------------- weight packing: W (N x K fp32) -> bf16 B-fragment tiles ----
// out tile (kt,nt): lane l holds col n = nt*16+(l&15), K = kt*32+((l>>4)<<3)+{0..7,16..23}
__global__ void k_pack(const float* __restrict__ W, u16* __restrict__ out,
                       int N, int K, int Ntiles, int Ktiles) {
  int gid = blockIdx.x * blockDim.x + threadIdx.x;
  int total = Ktiles * Ntiles * 32;
  if (gid >= total) return;
  int lane = gid & 31;
  int tile = gid >> 5;
  int kt = tile / Ntiles;
  int nt = tile - kt * Ntiles;
  int n  = nt * 16 + (lane & 15);
  int kb = kt * 32 + ((lane >> 4) << 3);
  u16* o = out + (size_t)tile * 512 + lane * 16;
  for (int j = 0; j < 16; ++j) {
    int k = kb + j + ((j >> 3) << 3);
    float v = (n < N && k < K) ? W[(size_t)n * K + k] : 0.f;
    o[j] = f2bf(v);
  }
}

// ---------------- state init ----------------
__global__ void k_init(const float* eh0, const float* ec0,
                       const float* mh0, const float* mc0,
                       u16* ench, float* encc, u16* hB, float* cS) {
  int i = blockIdx.x * blockDim.x + threadIdx.x;
  if (i < BATCH * ENC_H) { ench[i] = f2bf(eh0[i]); encc[i] = ec0[i]; }
  if (i < 4 * BATCH * MEM_H) { hB[i] = f2bf(mh0[i]); cS[i] = mc0[i]; }
}

// ---------------- convolutions (fp32 direct) ----------------
__global__ void k_conv1(const float* __restrict__ img, const float* __restrict__ w,
                        const float* __restrict__ b, float* __restrict__ out) {
  int o = blockIdx.x * blockDim.x + threadIdx.x;
  if (o >= 2048 * 16 * 11 * 11) return;
  int ox = o % 11, oy = (o / 11) % 11, oc = (o / 121) % 16, n = o / (121 * 16);
  float s = b[oc];
  for (int ic = 0; ic < 3; ++ic)
    for (int ky = 0; ky < 9; ++ky)
      for (int kx = 0; kx < 9; ++kx)
        s = fmaf(img[(((size_t)n * 3 + ic) * 99 + oy * 9 + ky) * 99 + ox * 9 + kx],
                 w[((oc * 3 + ic) * 9 + ky) * 9 + kx], s);
  out[o] = fmaxf(s, 0.f);
}
__global__ void k_conv2(const float* __restrict__ in, const float* __restrict__ w,
                        const float* __restrict__ b, float* __restrict__ out) {
  int o = blockIdx.x * blockDim.x + threadIdx.x;
  if (o >= 2048 * 32 * 9 * 9) return;
  int ox = o % 9, oy = (o / 9) % 9, oc = (o / 81) % 32, n = o / (81 * 32);
  float s = b[oc];
  for (int ic = 0; ic < 16; ++ic)
    for (int ky = 0; ky < 3; ++ky)
      for (int kx = 0; kx < 3; ++kx)
        s = fmaf(in[(((size_t)n * 16 + ic) * 11 + oy + ky) * 11 + ox + kx],
                 w[((oc * 16 + ic) * 3 + ky) * 3 + kx], s);
  out[o] = fmaxf(s, 0.f);
}
__global__ void k_conv3(const float* __restrict__ in, const float* __restrict__ w,
                        const float* __restrict__ b, u16* __restrict__ feats) {
  int o = blockIdx.x * blockDim.x + threadIdx.x;
  if (o >= 2048 * 32 * 7 * 7) return;
  int ox = o % 7, oy = (o / 7) % 7, oc = (o / 49) % 32, n = o / (49 * 32);
  float s = b[oc];
  for (int ic = 0; ic < 32; ++ic)
    for (int ky = 0; ky < 3; ++ky)
      for (int kx = 0; kx < 3; ++kx)
        s = fmaf(in[(((size_t)n * 32 + ic) * 9 + oy + ky) * 9 + ox + kx],
                 w[((oc * 32 + ic) * 3 + ky) * 3 + kx], s);
  feats[(size_t)n * 1568 + oc * 49 + oy * 7 + ox] = f2bf(fmaxf(s, 0.f));
}

// ---------------- FC: (2048x1568)@(1568x256)+b, relu -> hidden[:, :256] ------
__global__ void k_fc(const u16* __restrict__ feats, const u16* __restrict__ WP,
                     const float* __restrict__ bias, u16* __restrict__ hidden) {
  int lane = threadIdx.x & 31;
  int wid  = blockIdx.x * 8 + (threadIdx.x >> 5);   // 0..2047 tiles
  int r = wid >> 4, n = wid & 15;
  v8f acc = vzero8();
  for (int kt = 0; kt < 49; ++kt) {
    if (kt + 1 < 49)   // prefetch next A chunk (global_prefetch_b8)
      __builtin_prefetch((const void*)(feats + (size_t)(r * 16 + (lane & 15)) * 1568 +
                                       (kt + 1) * 32), 0, 3);
    v16bf a = frag_a(feats, 1568, r * 16, kt * 32, lane);
    acc = wmma_bf16(a, frag_b(WP, kt * 16 + n, lane), acc);
  }
  int col = n * 16 + (lane & 15);
  float bb = bias[col];
  int rsub = (lane >> 4) << 3;
  for (int e = 0; e < 8; ++e) {
    int row = r * 16 + e + rsub;
    hidden[(size_t)row * MEM_IN + col] = f2bf(fmaxf(acc[e] + bb, 0.f));
  }
}

// ---------------- enc LSTM: one step, one workgroup --------------------------
__global__ void k_enc_step(int t, const float* __restrict__ lang, const float* __restrict__ done,
                           u16* __restrict__ ench, float* __restrict__ encc,
                           const u16* __restrict__ WihP, const u16* __restrict__ WhhP,
                           const float* __restrict__ bih, const float* __restrict__ bhh,
                           u16* __restrict__ langH) {
  __shared__ __align__(16) u16 sX[BATCH * 32];
  __shared__ __align__(16) u16 sH[BATCH * ENC_H];
  // phase 1: stage padded x and masked h (bf16) into LDS
  for (int i = threadIdx.x; i < BATCH * 32; i += 256) {
    int b = i >> 5, k = i & 31;
    float v = (k < 14) ? lang[((size_t)t * BATCH + b) * 14 + k] : 0.f;
    sX[i] = f2bf(v);
  }
  for (int i = threadIdx.x; i < BATCH * ENC_H; i += 256) {
    int b = i >> 8;
    float m = 1.f - done[t * BATCH + b];
    sH[i] = f2bf(m * bf2f(ench[i]));
  }
  __syncthreads();
  // phase 2: gates GEMM via WMMA + fused cell. 32 (r,hcol) combos over 8 waves.
  int lane = threadIdx.x & 31;
  int wv   = threadIdx.x >> 5;
  for (int j = 0; j < 4; ++j) {
    int combo = wv + 8 * j;
    int r = combo >> 4, hcol = combo & 15;
    v8f aI = vzero8(), aF = vzero8(), aG = vzero8(), aO = vzero8();
    { // x part (K padded to 32)
      v16bf a = frag_a(sX, 32, r * 16, 0, lane);
      aI = wmma_bf16(a, frag_b(WihP,  0 + hcol, lane), aI);
      aF = wmma_bf16(a, frag_b(WihP, 16 + hcol, lane), aF);
      aG = wmma_bf16(a, frag_b(WihP, 32 + hcol, lane), aG);
      aO = wmma_bf16(a, frag_b(WihP, 48 + hcol, lane), aO);
    }
    for (int kt = 0; kt < 8; ++kt) { // h part (K=256)
      v16bf a = frag_a(sH, ENC_H, r * 16, kt * 32, lane);
      aI = wmma_bf16(a, frag_b(WhhP, kt * 64 +  0 + hcol, lane), aI);
      aF = wmma_bf16(a, frag_b(WhhP, kt * 64 + 16 + hcol, lane), aF);
      aG = wmma_bf16(a, frag_b(WhhP, kt * 64 + 32 + hcol, lane), aG);
      aO = wmma_bf16(a, frag_b(WhhP, kt * 64 + 48 + hcol, lane), aO);
    }
    int col = hcol * 16 + (lane & 15);
    float bI = bih[col]           + bhh[col];
    float bF = bih[ENC_H + col]   + bhh[ENC_H + col];
    float bG = bih[2*ENC_H + col] + bhh[2*ENC_H + col];
    float bO = bih[3*ENC_H + col] + bhh[3*ENC_H + col];
    int rsub = (lane >> 4) << 3;
    for (int e = 0; e < 8; ++e) {
      int b = r * 16 + e + rsub;
      float m  = 1.f - done[t * BATCH + b];
      float cm = m * encc[b * ENC_H + col];
      float cn = sigm(aF[e] + bF) * cm + sigm(aI[e] + bI) * tanhf(aG[e] + bG);
      float hn = sigm(aO[e] + bO) * tanhf(cn);
      encc[b * ENC_H + col] = cn;
      u16 hb = f2bf(hn);
      ench[b * ENC_H + col] = hb;
      langH[((size_t)(t * BATCH + b) << 8) + col] = hb;
    }
  }
}

// ---------------- emb: (2048x256)@(256x32)+b, relu -> hidden[:, 256:288] -----
__global__ void k_emb(const u16* __restrict__ langH, const u16* __restrict__ WP,
                      const float* __restrict__ bias, u16* __restrict__ hidden) {
  int lane = threadIdx.x & 31;
  int wid  = blockIdx.x * 8 + (threadIdx.x >> 5);   // 0..255 tiles
  int r = wid >> 1, n = wid & 1;
  v8f acc = vzero8();
  for (int kt = 0; kt < 8; ++kt) {
    v16bf a = frag_a(langH, ENC_H, r * 16, kt * 32, lane);
    acc = wmma_bf16(a, frag_b(WP, kt * 2 + n, lane), acc);
  }
  int col = n * 16 + (lane & 15);
  float bb = bias[col];
  int rsub = (lane >> 4) << 3;
  for (int e = 0; e < 8; ++e) {
    int row = r * 16 + e + rsub;
    hidden[(size_t)row * MEM_IN + ENC_H + col] = f2bf(fmaxf(acc[e] + bb, 0.f));
  }
}

// ---------------- mem LSTM: per-step mask h,c (hB -> hA masked) -------------
__global__ void k_mem_mask(int t, const float* __restrict__ done,
                           const u16* __restrict__ hB, u16* __restrict__ hA,
                           float* __restrict__ cS) {
  int i = blockIdx.x * blockDim.x + threadIdx.x;
  if (i >= 4 * BATCH * MEM_H) return;
  int b = (i >> 9) & 31;
  float m = 1.f - done[t * BATCH + b];
  hA[i] = f2bf(m * bf2f(hB[i]));
  cS[i] *= m;
}

// ---------------- mem LSTM layer: gates GEMM (WMMA) + fused cell ------------
// 64 waves: (r in 0..1) x (hcol in 0..31). Reads hOld (masked) + inp, writes hNew.
// Each workgroup's 8 waves share hcol octet h0..h0+7; B-fragments for the whole
// workgroup (4 gates x 8 tiles = 32KB per k-step) are staged into LDS by the
// Tensor Data Mover, double-buffered, synchronized with s_wait_tensorcnt.
__global__ void k_mem_layer(int t, int layer,
                            const u16* __restrict__ inp, int ldin, int kinTiles,
                            const u16* __restrict__ WihP, const u16* __restrict__ WhhP,
                            const u16* __restrict__ hOld, u16* __restrict__ hNew,
                            float* __restrict__ cS,
                            const float* __restrict__ bih, const float* __restrict__ bhh,
                            float* __restrict__ outBuf) {
  int lane = threadIdx.x & 31;
  int wv   = threadIdx.x >> 5;
  int wid  = blockIdx.x * 8 + wv;                   // 0..63
  int r = wid >> 5, hcol = wid & 31;
  v8f aI = vzero8(), aF = vzero8(), aG = vzero8(), aO = vzero8();
  const int KT = kinTiles + 16;

#if HAVE_TDM
  __shared__ __align__(16) u16 sW[2][4 * 8 * 512];  // 2 x 32KB double buffer
  int h0 = (blockIdx.x & 3) * 8;                    // workgroup's hcol base
  unsigned ldsBase = (unsigned)(uintptr_t)&sW[0][0];
  // slab base for k-step kt: Wih tiles first, then Whh tiles (same 128-tile layout)
  auto slabPtr = [&](int kt) -> const u16* {
    return (kt < kinTiles) ? (WihP + (size_t)kt * 65536 + (size_t)h0 * 512)
                           : (WhhP + (size_t)(kt - kinTiles) * 65536 + (size_t)h0 * 512);
  };
  if (wv == 0) {
    tdm_issue(slabPtr(0), ldsBase);
    tdm_issue(slabPtr(1), ldsBase + 32768u);
  }
  for (int kt = 0; kt < KT; ++kt) {
    if (wv == 0) {
      if (kt + 1 < KT) __builtin_amdgcn_s_wait_tensorcnt(1);
      else             __builtin_amdgcn_s_wait_tensorcnt(0);
    }
    __syncthreads();                                // buffer kt ready for all waves
    v16bf a = (kt < kinTiles)
                ? frag_a(inp,  ldin,  r * 16, kt * 32, lane)
                : frag_a(hOld, MEM_H, r * 16, (kt - kinTiles) * 32, lane);
    const u16* wb = &sW[kt & 1][0] + wv * 512 + lane * 16;
    aI = wmma_bf16(a, make_frag(wb,          wb + 8),          aI);
    aF = wmma_bf16(a, make_frag(wb + 4096,   wb + 4096 + 8),   aF);
    aG = wmma_bf16(a, make_frag(wb + 8192,   wb + 8192 + 8),   aG);
    aO = wmma_bf16(a, make_frag(wb + 12288,  wb + 12288 + 8),  aO);
    __syncthreads();                                // reads done; buffer reusable
    if (wv == 0 && kt + 2 < KT)
      tdm_issue(slabPtr(kt + 2), ldsBase + (unsigned)(kt & 1) * 32768u);
  }
#else
  for (int kt = 0; kt < kinTiles; ++kt) {           // input part
    v16bf a = frag_a(inp, ldin, r * 16, kt * 32, lane);
    aI = wmma_bf16(a, frag_b(WihP, kt * 128 +  0 + hcol, lane), aI);
    aF = wmma_bf16(a, frag_b(WihP, kt * 128 + 32 + hcol, lane), aF);
    aG = wmma_bf16(a, frag_b(WihP, kt * 128 + 64 + hcol, lane), aG);
    aO = wmma_bf16(a, frag_b(WihP, kt * 128 + 96 + hcol, lane), aO);
  }
  for (int kt = 0; kt < 16; ++kt) {                 // recurrent part (K=512)
    v16bf a = frag_a(hOld, MEM_H, r * 16, kt * 32, lane);
    aI = wmma_bf16(a, frag_b(WhhP, kt * 128 +  0 + hcol, lane), aI);
    aF = wmma_bf16(a, frag_b(WhhP, kt * 128 + 32 + hcol, lane), aF);
    aG = wmma_bf16(a, frag_b(WhhP, kt * 128 + 64 + hcol, lane), aG);
    aO = wmma_bf16(a, frag_b(WhhP, kt * 128 + 96 + hcol, lane), aO);
  }
#endif

  int col = hcol * 16 + (lane & 15);
  float bI = bih[col]           + bhh[col];
  float bF = bih[MEM_H + col]   + bhh[MEM_H + col];
  float bG = bih[2*MEM_H + col] + bhh[2*MEM_H + col];
  float bO = bih[3*MEM_H + col] + bhh[3*MEM_H + col];
  int rsub = (lane >> 4) << 3;
  for (int e = 0; e < 8; ++e) {
    int b = r * 16 + e + rsub;
    float cm = cS[b * MEM_H + col];                 // already masked
    float cn = sigm(aF[e] + bF) * cm + sigm(aI[e] + bI) * tanhf(aG[e] + bG);
    float hn = sigm(aO[e] + bO) * tanhf(cn);
    cS[b * MEM_H + col] = cn;
    hNew[b * MEM_H + col] = f2bf(hn);
    if (layer == 3) outBuf[((size_t)(t * BATCH + b) << 9) + col] = hn;
  }
}

// ---------------- heads: actor log-softmax + entropy + critic ---------------
__global__ void k_heads(const float* __restrict__ outBuf, const int* __restrict__ action,
                        const float* __restrict__ aw, const float* __restrict__ ab,
                        const float* __restrict__ cw, const float* __restrict__ cb,
                        float* __restrict__ out) {
  int tb = blockIdx.x * blockDim.x + threadIdx.x;
  if (tb >= 2048) return;
  const float* o = outBuf + (size_t)tb * MEM_H;
  float lg[NACT];
  for (int j = 0; j < NACT; ++j) lg[j] = ab[j];
  float val = cb[0];
  for (int k = 0; k < MEM_H; ++k) {
    float x = o[k];
#pragma unroll
    for (int j = 0; j < NACT; ++j) lg[j] = fmaf(x, aw[j * MEM_H + k], lg[j]);
    val = fmaf(x, cw[k], val);
  }
  float mx = lg[0];
  for (int j = 1; j < NACT; ++j) mx = fmaxf(mx, lg[j]);
  float s = 0.f;
  for (int j = 0; j < NACT; ++j) s += __expf(lg[j] - mx);
  float lse = mx + __logf(s);
  float ent = 0.f;
  for (int j = 0; j < NACT; ++j) { float lp = lg[j] - lse; ent -= __expf(lp) * lp; }
  int a = action[tb];
  out[tb * 3 + 0] = lg[a] - lse;
  out[tb * 3 + 1] = ent;
  out[tb * 3 + 2] = val;
}

// ---------------------------------------------------------------------------
extern "C" void kernel_launch(void* const* d_in, const int* in_sizes, int n_in,
                              void* d_out, int out_size, void* d_ws, size_t ws_size,
                              hipStream_t stream) {
  (void)in_sizes; (void)n_in; (void)out_size;
  const float* img     = (const float*)d_in[0];
  const float* lang    = (const float*)d_in[1];
  const float* done    = (const float*)d_in[2];
  const int*   action  = (const int*)  d_in[3];
  const float* enc_h0  = (const float*)d_in[4];
  const float* enc_c0  = (const float*)d_in[5];
  const float* mem_h0  = (const float*)d_in[6];
  const float* mem_c0  = (const float*)d_in[7];
  const float* conv1_w = (const float*)d_in[8];
  const float* conv1_b = (const float*)d_in[9];
  const float* conv2_w = (const float*)d_in[10];
  const float* conv2_b = (const float*)d_in[11];
  const float* conv3_w = (const float*)d_in[12];
  const float* conv3_b = (const float*)d_in[13];
  const float* fc_w    = (const float*)d_in[14];
  const float* fc_b    = (const float*)d_in[15];
  const float* enc_Wih = (const float*)d_in[16];
  const float* enc_Whh = (const float*)d_in[17];
  const float* enc_bih = (const float*)d_in[18];
  const float* enc_bhh = (const float*)d_in[19];
  const float* emb_w   = (const float*)d_in[20];
  const float* emb_b   = (const float*)d_in[21];
  const float* memWih0 = (const float*)d_in[22];
  const float* memWihR = (const float*)d_in[23];
  const float* memWhh  = (const float*)d_in[24];
  const float* mem_bih = (const float*)d_in[25];
  const float* mem_bhh = (const float*)d_in[26];
  const float* actor_w = (const float*)d_in[27];
  const float* actor_b = (const float*)d_in[28];
  const float* critic_w= (const float*)d_in[29];
  const float* critic_b= (const float*)d_in[30];
  float* dout = (float*)d_out;

  // ---- workspace layout ----
  char* ws = (char*)d_ws;
  size_t cur = 0;
  auto alloc = [&](size_t bytes) -> void* {
    size_t o = (cur + 255) & ~(size_t)255;
    cur = o + bytes;
    return (void*)(ws + o);
  };
  u16* pEncWih = (u16*)alloc((size_t)1  * 64  * 512 * 2);
  u16* pEncWhh = (u16*)alloc((size_t)8  * 64  * 512 * 2);
  u16* pEmb    = (u16*)alloc((size_t)8  * 2   * 512 * 2);
  u16* pFc     = (u16*)alloc((size_t)49 * 16  * 512 * 2);
  u16* pWih0   = (u16*)alloc((size_t)9  * 128 * 512 * 2);
  u16* pWihR   = (u16*)alloc((size_t)3 * 16 * 128 * 512 * 2);
  u16* pWhh    = (u16*)alloc((size_t)4 * 16 * 128 * 512 * 2);
  float* c1o   = (float*)alloc((size_t)2048 * 16 * 121 * 4);
  float* c2o   = (float*)alloc((size_t)2048 * 32 * 81  * 4);
  u16* feats   = (u16*)alloc((size_t)2048 * 1568 * 2);
  u16* hidden  = (u16*)alloc((size_t)2048 * MEM_IN * 2);
  u16* langH   = (u16*)alloc((size_t)2048 * ENC_H * 2);
  u16* ench    = (u16*)alloc((size_t)BATCH * ENC_H * 2);
  float* encc  = (float*)alloc((size_t)BATCH * ENC_H * 4);
  u16* hA      = (u16*)alloc((size_t)4 * BATCH * MEM_H * 2);   // masked states
  u16* hB      = (u16*)alloc((size_t)4 * BATCH * MEM_H * 2);   // fresh states
  float* cS    = (float*)alloc((size_t)4 * BATCH * MEM_H * 4);
  float* outB  = (float*)alloc((size_t)2048 * MEM_H * 4);
  if (cur > ws_size) return;

  auto grid1 = [](long long n) { return (unsigned)((n + 255) / 256); };

  // ---- pack weights into WMMA B-fragment layout (bf16) ----
  k_pack<<<grid1(1LL*64*32), 256, 0, stream>>>(enc_Wih, pEncWih, 1024, 14, 64, 1);
  k_pack<<<grid1(8LL*64*32), 256, 0, stream>>>(enc_Whh, pEncWhh, 1024, 256, 64, 8);
  k_pack<<<grid1(8LL*2*32),  256, 0, stream>>>(emb_w,   pEmb,    32,   256, 2,  8);
  k_pack<<<grid1(49LL*16*32),256, 0, stream>>>(fc_w,    pFc,     256, 1568, 16, 49);
  k_pack<<<grid1(9LL*128*32),256, 0, stream>>>(memWih0, pWih0,  2048,  288, 128, 9);
  for (int l = 0; l < 3; ++l)
    k_pack<<<grid1(16LL*128*32), 256, 0, stream>>>(memWihR + (size_t)l*2048*512,
        pWihR + (size_t)l*16*128*512, 2048, 512, 128, 16);
  for (int l = 0; l < 4; ++l)
    k_pack<<<grid1(16LL*128*32), 256, 0, stream>>>(memWhh + (size_t)l*2048*512,
        pWhh + (size_t)l*16*128*512, 2048, 512, 128, 16);

  // ---- init recurrent states ----
  k_init<<<grid1(4LL*BATCH*MEM_H), 256, 0, stream>>>(enc_h0, enc_c0, mem_h0, mem_c0,
                                                     ench, encc, hB, cS);

  // ---- image encoder ----
  k_conv1<<<grid1(2048LL*16*121), 256, 0, stream>>>(img, conv1_w, conv1_b, c1o);
  k_conv2<<<grid1(2048LL*32*81),  256, 0, stream>>>(c1o, conv2_w, conv2_b, c2o);
  k_conv3<<<grid1(2048LL*32*49),  256, 0, stream>>>(c2o, conv3_w, conv3_b, feats);
  k_fc<<<256, 256, 0, stream>>>(feats, pFc, fc_b, hidden);

  // ---- language encoder (sequential, WMMA per step) ----
  for (int t = 0; t < T_STEPS; ++t)
    k_enc_step<<<1, 256, 0, stream>>>(t, lang, done, ench, encc,
                                      pEncWih, pEncWhh, enc_bih, enc_bhh, langH);
  k_emb<<<32, 256, 0, stream>>>(langH, pEmb, emb_b, hidden);

  // ---- 4-layer memory LSTM (sequential steps, TDM-staged WMMA gates) ----
  for (int t = 0; t < T_STEPS; ++t) {
    k_mem_mask<<<grid1(4LL*BATCH*MEM_H), 256, 0, stream>>>(t, done, hB, hA, cS);
    for (int l = 0; l < 4; ++l) {
      const u16* inp; int ldin, kinT; const u16* WP;
      if (l == 0) { inp = hidden + (size_t)t*BATCH*MEM_IN; ldin = MEM_IN; kinT = 9;  WP = pWih0; }
      else        { inp = hB + (size_t)(l-1)*BATCH*MEM_H;  ldin = MEM_H;  kinT = 16; WP = pWihR + (size_t)(l-1)*16*128*512; }
      k_mem_layer<<<8, 256, 0, stream>>>(t, l, inp, ldin, kinT, WP,
          pWhh + (size_t)l*16*128*512,
          hA + (size_t)l*BATCH*MEM_H, hB + (size_t)l*BATCH*MEM_H,
          cS + (size_t)l*BATCH*MEM_H,
          mem_bih + (size_t)l*2048, mem_bhh + (size_t)l*2048, outB);
    }
  }

  // ---- heads ----
  k_heads<<<8, 256, 0, stream>>>(outB, action, actor_w, actor_b, critic_w, critic_b, dout);
}
